// GATModel_55061480734921
// MI455X (gfx1250) — matrix operand order repaired and checked by reading, not compiled, not run on hardware
//
#include <hip/hip_runtime.h>
#include <hip/hip_bf16.h>

#define NN   20000
#define EE   320000
#define DIN  128
#define HID  64
#define NH   4
#define BB   16
#define NEG_SLOPE 0.2f

typedef __attribute__((ext_vector_type(16))) _Float16 v16h;
typedef __attribute__((ext_vector_type(8)))  float    v8f;

// ---------- helpers ----------
__device__ __forceinline__ unsigned ordf(float f) {
    unsigned u = __float_as_uint(f);
    return (u & 0x80000000u) ? ~u : (u | 0x80000000u);
}
__device__ __forceinline__ float unordf(unsigned u) {
    u = (u & 0x80000000u) ? (u & 0x7fffffffu) : ~u;
    return __uint_as_float(u);
}

__global__ void fill_f32(float* p, float v, int n) {
    int i = blockIdx.x * blockDim.x + threadIdx.x;
    if (i < n) p[i] = v;
}
__global__ void fill_u32(unsigned* p, unsigned v, int n) {
    int i = blockIdx.x * blockDim.x + threadIdx.x;
    if (i < n) p[i] = v;
}

// ---------- WMMA strip GEMM: C[M,NOUT] = A[M,K] @ W[K,NOUT] ----------
// f32 in memory, f16 MACs, f32 accumulate. One wave per block; each wave owns a
// 16x64 output strip: one shared A fragment, 4 B fragments, 4 WMMAs per k-step.
// Requires M%16==0, K%32==0, NOUT%64==0. Fully unrolled (K, NOUT compile-time).
template<int K, int NOUT>
__global__ void __launch_bounds__(32)
gemm_wmma_strip(const float* __restrict__ A, const float* __restrict__ W,
                float* __restrict__ Cout)
{
    const int lane = threadIdx.x;        // 0..31, full wave always active
    const int g    = lane >> 4;          // half-wave select
    const int m    = lane & 15;          // row (A) / col (B,C)

    const int stripsN = NOUT / 64;
    const int strip   = blockIdx.x;
    const int tm      = strip / stripsN;
    const int cb      = (strip - tm * stripsN) * 64;   // strip column base

    const float* arow = A + (size_t)(tm * 16 + m) * K;

    const v8f zero = {0.f, 0.f, 0.f, 0.f, 0.f, 0.f, 0.f, 0.f};
    v8f acc[4] = {zero, zero, zero, zero};

#pragma unroll
    for (int k0 = 0; k0 < K; k0 += 32) {
        // A 16x32 f16 fragment: halves 0..7 -> K=k0+8g+j ; 8..15 -> K=k0+16+8g+(j-8)
        // Two 32B-aligned float4 pairs per lane.
        const float4* p0 = (const float4*)(arow + k0 + 8 * g);
        const float4* p1 = (const float4*)(arow + k0 + 16 + 8 * g);
        float4 q0 = p0[0], q1 = p0[1], q2 = p1[0], q3 = p1[1];
        float av[16] = {q0.x, q0.y, q0.z, q0.w, q1.x, q1.y, q1.z, q1.w,
                        q2.x, q2.y, q2.z, q2.w, q3.x, q3.y, q3.z, q3.w};
        v16h a;
#pragma unroll
        for (int j = 0; j < 16; ++j) a[j] = (_Float16)av[j];

        // B 32x16 f16 fragments: halves 0..7 -> K=k0+8g+j ; 8..15 -> K=k0+16+8g+(j-8), col = cb + 16t + m
        const float* wbase = W + (size_t)(k0 + 8 * g) * NOUT + cb + m;
#pragma unroll
        for (int t = 0; t < 4; ++t) {
            v16h b;
#pragma unroll
            for (int j = 0; j < 16; ++j)
                b[j] = (_Float16)wbase[(size_t)(j + (j & 8)) * NOUT + t * 16];
            acc[t] = __builtin_amdgcn_wmma_f32_16x16x32_f16(
                false, a, false, b, (short)0, acc[t], false, false);
        }
    }
    // C 16x16 f32 tiles: VGPR r -> row tm*16 + r + 8g, col cb + 16t + m
    float* crow = Cout + (size_t)(tm * 16 + 8 * g) * NOUT + cb + m;
#pragma unroll
    for (int r = 0; r < 8; ++r) {
#pragma unroll
        for (int t = 0; t < 4; ++t)
            crow[(size_t)r * NOUT + t * 16] = acc[t][r];
    }
}

// ---------- per-(node,head) attention coefficients ----------
__global__ void attn_coef(const float* __restrict__ h, const float* __restrict__ asrc,
                          const float* __restrict__ adst, float* __restrict__ esrc,
                          float* __restrict__ edst, int Hh, int C, int n)
{
    int t = blockIdx.x * blockDim.x + threadIdx.x;
    if (t >= n * Hh) return;
    int node = t / Hh, hd = t - node * Hh;
    const float* row = h + (size_t)node * Hh * C + hd * C;
    const float* a1 = asrc + hd * C;
    const float* a2 = adst + hd * C;
    float s1 = 0.f, s2 = 0.f;
    for (int c = 0; c < C; ++c) { float v = row[c]; s1 += v * a1[c]; s2 += v * a2[c]; }
    esrc[t] = s1;
    edst[t] = s2;
}

// ---------- pass A: leaky_relu logits + segment max (ordered-uint atomicMax) ----------
__global__ void edge_logits(const int* __restrict__ ei, int Hh,
                            const float* __restrict__ esrc, const float* __restrict__ edst,
                            float* __restrict__ ew, unsigned* __restrict__ mbuf)
{
    const int Etot = EE + NN;
    int t = blockIdx.x * blockDim.x + threadIdx.x;
    if (t >= Etot * Hh) return;
    int e = t / Hh, hd = t - e * Hh;
    int s = (e < EE) ? ei[e]      : (e - EE);
    int d = (e < EE) ? ei[EE + e] : (e - EE);
    float v = esrc[s * Hh + hd] + edst[d * Hh + hd];
    v = (v > 0.f) ? v : NEG_SLOPE * v;
    ew[t] = v;
    atomicMax(&mbuf[d * Hh + hd], ordf(v));
}

// ---------- pass B: exp(e - max) + segment sum ----------
__global__ void edge_exp(const int* __restrict__ ei, int Hh,
                         float* __restrict__ ew, const unsigned* __restrict__ mbuf,
                         float* __restrict__ den)
{
    const int Etot = EE + NN;
    int t = blockIdx.x * blockDim.x + threadIdx.x;
    if (t >= Etot * Hh) return;
    int e = t / Hh, hd = t - e * Hh;
    int d = (e < EE) ? ei[EE + e] : (e - EE);
    float w = expf(ew[t] - unordf(mbuf[d * Hh + hd]));
    ew[t] = w;
    atomicAdd(&den[d * Hh + hd], w);
}

// ---------- pass C: weighted scatter-add of source features ----------
// one block per edge, blockDim = Hh*C (256 or 64); head uniform within a wave
__global__ void edge_aggregate(const int* __restrict__ ei, int Hh, int C,
                               const float* __restrict__ hfeat, const float* __restrict__ ew,
                               const float* __restrict__ den, float* __restrict__ agg)
{
    int e = blockIdx.x;
    int t = threadIdx.x;
    int s = (e < EE) ? ei[e]      : (e - EE);
    int d = (e < EE) ? ei[EE + e] : (e - EE);
    int hd = t / C;
    float alpha = ew[(size_t)e * Hh + hd] / den[d * Hh + hd];
    int HC = Hh * C;
    atomicAdd(&agg[(size_t)d * HC + t], alpha * hfeat[(size_t)s * HC + t]);
}

// ---------- out = elu(out + bias) elementwise ----------
__global__ void bias_elu(float* __restrict__ buf, const float* __restrict__ bias, int n, int w)
{
    int i = blockIdx.x * blockDim.x + threadIdx.x;
    if (i >= n * w) return;
    float v = buf[i] + bias[i % w];
    buf[i] = (v > 0.f) ? v : (expf(v) - 1.f);
}

// ---------- attention pooling ----------
__global__ void pool_attn(const float* __restrict__ h, const float* __restrict__ wp,
                          const float* __restrict__ bp, const int* __restrict__ batch,
                          float* __restrict__ sums, float* __restrict__ counts)
{
    int i = blockIdx.x * blockDim.x + threadIdx.x;
    if (i >= NN) return;
    const float* row = h + (size_t)i * HID;
    float acc = bp[0];
    for (int c = 0; c < HID; ++c) acc += row[c] * wp[c];
    float a = 1.f / (1.f + expf(-acc));
    int b = batch[i];
    for (int c = 0; c < HID; ++c) atomicAdd(&sums[b * HID + c], row[c] * a);
    atomicAdd(&counts[b], 1.f);
}

__global__ void final_reg(const float* __restrict__ sums, const float* __restrict__ counts,
                          const float* __restrict__ wr, const float* __restrict__ br,
                          float* __restrict__ out)
{
    int b = threadIdx.x;
    if (b >= BB) return;
    float cnt = fmaxf(counts[b], 1.f);
    float acc = 0.f;
    for (int c = 0; c < HID; ++c) acc += (sums[b * HID + c] / cnt) * wr[c];
    out[b] = acc + br[0];
}

// ---------- host-side layer driver ----------
template<int K, int Hh, int C>
static void run_gat_layer(const float* A,
                          const float* W, const float* asrc, const float* adst, const float* bias,
                          const int* ei, float* hfeat, float* agg,
                          float* esrc, float* edst, unsigned* mbuf, float* den, float* ew,
                          hipStream_t stream)
{
    constexpr int HC = Hh * C;
    const int nh   = NN * Hh;
    const int Etot = EE + NN;
    const int EH   = Etot * Hh;

    gemm_wmma_strip<K, HC><<<(NN / 16) * (HC / 64), 32, 0, stream>>>(A, W, hfeat);

    attn_coef<<<(nh + 255) / 256, 256, 0, stream>>>(hfeat, asrc, adst, esrc, edst, Hh, C, NN);

    fill_u32<<<(nh + 255) / 256, 256, 0, stream>>>(mbuf, 0u, nh);
    fill_f32<<<(nh + 255) / 256, 256, 0, stream>>>(den, 0.f, nh);

    edge_logits<<<(EH + 255) / 256, 256, 0, stream>>>(ei, Hh, esrc, edst, ew, mbuf);
    edge_exp<<<(EH + 255) / 256, 256, 0, stream>>>(ei, Hh, ew, mbuf, den);

    fill_f32<<<(NN * HC + 255) / 256, 256, 0, stream>>>(agg, 0.f, NN * HC);
    edge_aggregate<<<Etot, HC, 0, stream>>>(ei, Hh, C, hfeat, ew, den, agg);

    bias_elu<<<(NN * HC + 255) / 256, 256, 0, stream>>>(agg, bias, NN, HC);
}

extern "C" void kernel_launch(void* const* d_in, const int* in_sizes, int n_in,
                              void* d_out, int out_size, void* d_ws, size_t ws_size,
                              hipStream_t stream)
{
    (void)in_sizes; (void)n_in; (void)out_size; (void)ws_size;

    const float* x    = (const float*)d_in[0];
    const int*   ei   = (const int*)  d_in[1];
    const int*   bat  = (const int*)  d_in[2];
    const float* W1   = (const float*)d_in[3];
    const float* as1  = (const float*)d_in[4];
    const float* ad1  = (const float*)d_in[5];
    const float* b1   = (const float*)d_in[6];
    const float* W2   = (const float*)d_in[7];
    const float* as2  = (const float*)d_in[8];
    const float* ad2  = (const float*)d_in[9];
    const float* b2   = (const float*)d_in[10];
    const float* W3   = (const float*)d_in[11];
    const float* as3  = (const float*)d_in[12];
    const float* ad3  = (const float*)d_in[13];
    const float* b3   = (const float*)d_in[14];
    const float* wp   = (const float*)d_in[15];
    const float* bp   = (const float*)d_in[16];
    const float* wr   = (const float*)d_in[17];
    const float* br   = (const float*)d_in[18];
    float* out        = (float*)d_out;

    // workspace layout (floats)
    float* ws = (float*)d_ws;
    size_t off = 0;
    float*    bufA   = ws + off; off += (size_t)NN * 256;          // GEMM output / hfeat
    float*    bufB   = ws + off; off += (size_t)NN * 256;          // agg / next-layer input
    float*    esrc   = ws + off; off += (size_t)NN * NH;
    float*    edst   = ws + off; off += (size_t)NN * NH;
    float*    den    = ws + off; off += (size_t)NN * NH;
    unsigned* mbuf   = (unsigned*)(ws + off); off += (size_t)NN * NH;
    float*    ew     = ws + off; off += (size_t)(EE + NN) * NH;
    float*    sums   = ws + off; off += (size_t)BB * HID;
    float*    counts = ws + off; off += (size_t)BB;

    // layer 1: DIN -> (4,64) concat ; input x
    run_gat_layer<DIN, NH, HID>(x, W1, as1, ad1, b1, ei, bufA, bufB,
                                esrc, edst, mbuf, den, ew, stream);
    // layer 2: 256 -> (4,64) concat ; GEMM consumes bufB before agg overwrites it
    run_gat_layer<NH * HID, NH, HID>(bufB, W2, as2, ad2, b2, ei, bufA, bufB,
                                     esrc, edst, mbuf, den, ew, stream);
    // layer 3: 256 -> (1,64), mean over 1 head == identity
    run_gat_layer<NH * HID, 1, HID>(bufB, W3, as3, ad3, b3, ei, bufA, bufB,
                                    esrc, edst, mbuf, den, ew, stream);

    // attention pooling + regressor
    fill_f32<<<(BB * HID + 255) / 256, 256, 0, stream>>>(sums, 0.f, BB * HID);
    fill_f32<<<1, 64, 0, stream>>>(counts, 0.f, BB);
    pool_attn<<<(NN + 255) / 256, 256, 0, stream>>>(bufB, wp, bp, bat, sums, counts);
    final_reg<<<1, 32, 0, stream>>>(sums, counts, wr, br, out);
}
